// RelationMemory_21801253995008
// MI455X (gfx1250) — matrix operand order repaired and checked by reading, not compiled, not run on hardware
//
#include <hip/hip_runtime.h>
#include <hip/hip_bf16.h>
#include <cmath>

// Problem constants (match reference)
constexpr int kB  = 64;
constexpr int kD  = 128;
constexpr int kK1 = 4096;
constexpr int kN  = 100000;
constexpr float kInvT = 1.0f / 0.07f;

typedef __attribute__((ext_vector_type(16))) __bf16 v16bf;
typedef __attribute__((ext_vector_type(8)))  float  v8f;

constexpr int WLDW = 136;                 // padded row width (bf16) for 128-wide tiles
constexpr int ELDW = 132;                 // padded row width (fp32) for the et tile
constexpr int NW   = 4;                   // waves per block in fused kernel
constexpr int NTILES = (kK1 * kB) / 16;   // 16384 row tiles

// Per-wave scratch (ushort units): xs, ys1, ys2 (bf16 tiles) + et (fp32 tile)
constexpr int PW_TILE  = 16 * WLDW;                  // one bf16 tile
constexpr int PW_ET    = 16 * ELDW * 2;              // fp32 tile, in ushort units
constexpr int PW_SIZE  = 3 * PW_TILE + PW_ET;        // 10752 ushorts per wave

// LDS layout (ushort units) for the fused kernel
constexpr int OFF_W    = 0;                          // 6 * 128 * WLDW transposed bf16 weights
constexpr int OFF_PW   = 6 * kD * WLDW;              // NW * PW_SIZE per-wave tiles
constexpr int OFF_BIAS = OFF_PW + NW * PW_SIZE;      // 6 * 128 fp32 biases
constexpr int SMEM_USH = OFF_BIAS + 6 * kD * 2;      // = 148992 ushorts = 297984 B

// ---------------------------------------------------------------------------
// Truncating fp32 -> bf16 (store-side folds to d16_hi moves/stores; pack is a
// cheap bitfield merge the compiler can lower to v_perm_b32).
__device__ __forceinline__ unsigned short f2bf(float f) {
  return (unsigned short)(__float_as_uint(f) >> 16);
}
__device__ __forceinline__ unsigned f2bf_pk(float a, float b) {
  return (__float_as_uint(a) >> 16) | (__float_as_uint(b) & 0xFFFF0000u);
}
__device__ __forceinline__ float bf2f(unsigned short u) {
  return __uint_as_float(((unsigned)u) << 16);
}

union FragCvt { v16bf v; uint4 q[2]; };

// A-operand (16x32 bf16) from row-major [16][WLDW] bf16 LDS tile.
// lane: m = lane&15 -> row M; h = lane>>4 selects K-half.
//   elems 0..7  -> K = kc*32 + 8h + 0..7
//   elems 8..15 -> K = kc*32 + 16 + 8h + 0..7
__device__ __forceinline__ v16bf load_afrag(const unsigned short* tile, int m, int h, int kc) {
  FragCvt f;
  const unsigned short* r = tile + m * WLDW + kc * 32 + 8 * h;
  f.q[0] = *(const uint4*)(r);
  f.q[1] = *(const uint4*)(r + 16);
  return f.v;
}

// B-operand (32x16 bf16) from weight LDS [e][d]:
// lane L -> N = nt*16 + (L&15); elems 0..7 -> K = kc*32 + 8h + 0..7, elems 8..15 -> K = kc*32 + 16 + 8h + 0..7.
__device__ __forceinline__ v16bf load_bfrag(const unsigned short* wT, int lane, int kc, int nt) {
  FragCvt f;
  const unsigned short* r = wT + (nt * 16 + (lane & 15)) * WLDW + kc * 32 + 8 * (lane >> 4);
  f.q[0] = *(const uint4*)(r);
  f.q[1] = *(const uint4*)(r + 16);
  return f.v;
}

// ---------------------------------------------------------------------------
// Kernel 1: a2[br][b][e] = (br?v1:v2)[b] @ W1^T + b1   (tiny: 2*64*128 outs)
__global__ void a2_kernel(const float* __restrict__ v1, const float* __restrict__ v2,
                          const float* __restrict__ mt_w1, const float* __restrict__ mt_b1,
                          const float* __restrict__ mts_w1, const float* __restrict__ mts_b1,
                          float* __restrict__ a2) {
  int b = blockIdx.x;
  int tid = threadIdx.x;        // 256 threads
  int br = tid >> 7;            // 0 = t branch (v2,mt), 1 = ts branch (v1,mts)
  int e = tid & 127;
  const float* x  = (br ? v1 : v2) + b * kD;
  const float* w  = (br ? mts_w1 : mt_w1) + e * kD;
  const float* bb = br ? mts_b1 : mt_b1;
  float s = bb[e];
  for (int d = 0; d < kD; ++d) s += x[d] * w[d];
  a2[br * kB * kD + b * kD + e] = s;
}

// Kernel 2: bulk copy memory bank into output (new_memory base image)
__global__ void copy_mem_kernel(const float4* __restrict__ src, float4* __restrict__ dst, int n4) {
  int i = blockIdx.x * blockDim.x + threadIdx.x;
  if (i < n4) dst[i] = src[i];
}

// Kernel 3: momentum update + l2norm of the 64 touched rows
__global__ void update_mem_kernel(const float* __restrict__ mem, const float* __restrict__ v2,
                                  const int* __restrict__ y, float* __restrict__ out_mem) {
  int b = blockIdx.x;           // 64 blocks, 32 lanes (one wave per row)
  int lane = threadIdx.x;
  int row = y[b];
  float4 mv = ((const float4*)(mem + (long)row * kD))[lane];
  float4 vv = ((const float4*)(v2 + (long)b * kD))[lane];
  float4 ab;
  ab.x = 0.5f * (mv.x + vv.x); ab.y = 0.5f * (mv.y + vv.y);
  ab.z = 0.5f * (mv.z + vv.z); ab.w = 0.5f * (mv.w + vv.w);
  float ss = ab.x*ab.x + ab.y*ab.y + ab.z*ab.z + ab.w*ab.w;
  for (int off = 1; off < 32; off <<= 1) ss += __shfl_xor(ss, off);
  float inv = rsqrtf(ss);
  ab.x *= inv; ab.y *= inv; ab.z *= inv; ab.w *= inv;
  ((float4*)(out_mem + (long)row * kD))[lane] = ab;
}

// ---------------------------------------------------------------------------
// Kernel 4: fused gather + 6-GEMM chain (bf16 WMMA) + relu + l2norm + score
__global__ void __launch_bounds__(NW * 32) fused_kernel(
    const float* __restrict__ mem, const int* __restrict__ idx,
    const float* __restrict__ a2,
    const float* __restrict__ mt_w2,  const float* __restrict__ mt_b2,
    const float* __restrict__ mt_wv,  const float* __restrict__ mt_bv,
    const float* __restrict__ ht_w,   const float* __restrict__ ht_b,
    const float* __restrict__ mts_w2, const float* __restrict__ mts_b2,
    const float* __restrict__ mts_wv, const float* __restrict__ mts_bv,
    const float* __restrict__ hts_w,  const float* __restrict__ hts_b,
    float* __restrict__ score) {
  extern __shared__ unsigned short sm[];
  unsigned short* wls = sm + OFF_W;               // transposed bf16 weights
  unsigned short* pw  = sm + OFF_PW;              // per-wave tile scratch
  float*          bl  = (float*)(sm + OFF_BIAS);  // fp32 biases

  const int lane = threadIdx.x & 31;
  const int wave = threadIdx.x >> 5;
  const int m = lane & 15, h = lane >> 4;

  // One-time block staging: weights (transposed, bf16) + biases (fp32).
  {
    const float* Wg[6] = {mt_w2, mt_wv, ht_w, mts_w2, mts_wv, hts_w};
    for (int w = 0; w < 6; ++w) {
      const float* src = Wg[w];
      unsigned short* dst = wls + w * kD * WLDW;
      for (int i = threadIdx.x; i < kD * kD; i += blockDim.x) {
        int e = i >> 7, d = i & 127;
        dst[e * WLDW + d] = f2bf(src[i]);
      }
    }
    const float* Bg[6] = {mt_b2, mt_bv, ht_b, mts_b2, mts_bv, hts_b};
    for (int w = 0; w < 6; ++w)
      for (int i = threadIdx.x; i < kD; i += blockDim.x)
        bl[w * kD + i] = Bg[w][i];
  }
  __syncthreads();

  unsigned short* xs  = pw + wave * PW_SIZE;      // gathered memory rows (bf16)
  unsigned short* ys1 = xs  + PW_TILE;            // pre = relu(a - b)     (bf16)
  unsigned short* ys2 = ys1 + PW_TILE;            // h   = pre@Wv^T + bv   (bf16)
  float*          et  = (float*)(ys2 + PW_TILE);  // normalized t-branch embedding (fp32)

  for (int t = blockIdx.x * NW + wave; t < NTILES; t += gridDim.x * NW) {
    const int s  = t >> 2;          // 64/16 = 4 row-tiles per sample index s
    const int b0 = (t & 3) << 4;

    // Gather 16 memory rows for this tile into LDS (fp32 -> bf16, b64 stores).
    {
      int g = idx[(b0 + m) * kK1 + s];
      const float4* srow = ((const float4*)(mem + (long)g * kD)) + h * 16;
      uint2* xr = (uint2*)(xs + m * WLDW + h * 64);
#pragma unroll
      for (int j = 0; j < 16; ++j) {
        float4 f = srow[j];
        uint2 p;
        p.x = f2bf_pk(f.x, f.y);
        p.y = f2bf_pk(f.z, f.w);
        xr[j] = p;
      }
    }

    // Software prefetch of the next tile's gather rows (global_prefetch_b8).
    {
      int t2 = t + gridDim.x * NW;
      if (t2 < NTILES) {
        int s2 = t2 >> 2, b02 = (t2 & 3) << 4;
        int g2 = idx[(b02 + m) * kK1 + s2];
        const char* p = (const char*)(mem + (long)g2 * kD) + h * 256;
        __builtin_prefetch(p, 0, 1);
        __builtin_prefetch(p + 128, 0, 1);
      }
    }

    float dacc[8];
#pragma unroll
    for (int v = 0; v < 8; ++v) dacc[v] = 0.f;

    for (int br = 0; br < 2; ++br) {
      const unsigned short* w2l = wls + (br * 3 + 0) * kD * WLDW;
      const unsigned short* wvl = wls + (br * 3 + 1) * kD * WLDW;
      const unsigned short* hwl = wls + (br * 3 + 2) * kD * WLDW;
      const float* b2p = bl + (br * 3 + 0) * kD;
      const float* bvp = bl + (br * 3 + 1) * kD;
      const float* hbp = bl + (br * 3 + 2) * kD;
      const float* a2p = a2 + br * kB * kD;       // fp32, L2-resident

      // GEMM1: pre = relu(a2 - (X @ W2^T + b2)) -> ys1
      {
        v16bf af[4];
#pragma unroll
        for (int kc = 0; kc < 4; ++kc) af[kc] = load_afrag(xs, m, h, kc);
#pragma unroll
        for (int nt = 0; nt < 8; ++nt) {
          v8f acc = {0.f,0.f,0.f,0.f,0.f,0.f,0.f,0.f};
#pragma unroll
          for (int kc = 0; kc < 4; ++kc)
            acc = __builtin_amdgcn_wmma_f32_16x16x32_bf16(false, af[kc], false,
                    load_bfrag(w2l, lane, kc, nt), (short)0, acc, false, false);
          int e = nt * 16 + m;
          float bias = b2p[e];
#pragma unroll
          for (int v = 0; v < 8; ++v) {
            int row = v + 8 * h;
            float pre = a2p[(b0 + row) * kD + e] - (acc[v] + bias);
            ys1[row * WLDW + e] = f2bf(fmaxf(pre, 0.f));
          }
        }
      }
      // GEMM2: h = pre @ Wv^T + bv -> ys2
      {
        v16bf af[4];
#pragma unroll
        for (int kc = 0; kc < 4; ++kc) af[kc] = load_afrag(ys1, m, h, kc);
#pragma unroll
        for (int nt = 0; nt < 8; ++nt) {
          v8f acc = {0.f,0.f,0.f,0.f,0.f,0.f,0.f,0.f};
#pragma unroll
          for (int kc = 0; kc < 4; ++kc)
            acc = __builtin_amdgcn_wmma_f32_16x16x32_bf16(false, af[kc], false,
                    load_bfrag(wvl, lane, kc, nt), (short)0, acc, false, false);
          int e = nt * 16 + m;
          float bias = bvp[e];
#pragma unroll
          for (int v = 0; v < 8; ++v)
            ys2[(v + 8 * h) * WLDW + e] = f2bf(acc[v] + bias);
        }
      }
      // GEMM3: emb = h @ Hw^T + hb; keep in regs; l2-normalize; fold into dot.
      {
        v16bf af[4];
#pragma unroll
        for (int kc = 0; kc < 4; ++kc) af[kc] = load_afrag(ys2, m, h, kc);
        float o[8][8];
        float sq[8];
#pragma unroll
        for (int v = 0; v < 8; ++v) sq[v] = 0.f;
#pragma unroll
        for (int nt = 0; nt < 8; ++nt) {
          v8f acc = {0.f,0.f,0.f,0.f,0.f,0.f,0.f,0.f};
#pragma unroll
          for (int kc = 0; kc < 4; ++kc)
            acc = __builtin_amdgcn_wmma_f32_16x16x32_bf16(false, af[kc], false,
                    load_bfrag(hwl, lane, kc, nt), (short)0, acc, false, false);
          float bias = hbp[nt * 16 + m];
#pragma unroll
          for (int v = 0; v < 8; ++v) {
            float val = acc[v] + bias;
            o[nt][v] = val;
            sq[v] += val * val;
          }
        }
        // row-norm: reduce over the 16 lanes of each half (cols of one row)
#pragma unroll
        for (int off = 1; off < 16; off <<= 1)
#pragma unroll
          for (int v = 0; v < 8; ++v) sq[v] += __shfl_xor(sq[v], off);
        float inv[8];
#pragma unroll
        for (int v = 0; v < 8; ++v) inv[v] = rsqrtf(sq[v]);
        if (br == 0) {
#pragma unroll
          for (int nt = 0; nt < 8; ++nt)
#pragma unroll
            for (int v = 0; v < 8; ++v)
              et[(v + 8 * h) * ELDW + nt * 16 + m] = o[nt][v] * inv[v];
        } else {
#pragma unroll
          for (int nt = 0; nt < 8; ++nt)
#pragma unroll
            for (int v = 0; v < 8; ++v)
              dacc[v] += (o[nt][v] * inv[v]) * et[(v + 8 * h) * ELDW + nt * 16 + m];
        }
      }
    } // branch

    // dot reduce across the 16 column-lanes; two lanes (m==0, h=0/1) emit 8 rows each
#pragma unroll
    for (int off = 1; off < 16; off <<= 1)
#pragma unroll
      for (int v = 0; v < 8; ++v) dacc[v] += __shfl_xor(dacc[v], off);
    if (m == 0) {
#pragma unroll
      for (int v = 0; v < 8; ++v) {
        int r = t * 16 + v + 8 * h;           // == s*64 + b
        score[r] = __expf((dacc[v] - 1.0f) * kInvT);
      }
    }
  }
}

// ---------------------------------------------------------------------------
extern "C" void kernel_launch(void* const* d_in, const int* in_sizes, int n_in,
                              void* d_out, int out_size, void* d_ws, size_t ws_size,
                              hipStream_t stream) {
  const float* v1     = (const float*)d_in[0];
  const float* v2     = (const float*)d_in[1];
  const float* mem    = (const float*)d_in[2];
  const float* mt_w1  = (const float*)d_in[3];
  const float* mt_b1  = (const float*)d_in[4];
  const float* mt_w2  = (const float*)d_in[5];
  const float* mt_b2  = (const float*)d_in[6];
  const float* mt_wv  = (const float*)d_in[7];
  const float* mt_bv  = (const float*)d_in[8];
  const float* mts_w1 = (const float*)d_in[9];
  const float* mts_b1 = (const float*)d_in[10];
  const float* mts_w2 = (const float*)d_in[11];
  const float* mts_b2 = (const float*)d_in[12];
  const float* mts_wv = (const float*)d_in[13];
  const float* mts_bv = (const float*)d_in[14];
  const float* ht_w   = (const float*)d_in[15];
  const float* ht_b   = (const float*)d_in[16];
  const float* hts_w  = (const float*)d_in[17];
  const float* hts_b  = (const float*)d_in[18];
  const int*   y      = (const int*)d_in[19];
  const int*   idx    = (const int*)d_in[20];

  float* out_score = (float*)d_out;                        // [K1*B] = 262144
  float* out_mem   = out_score + (size_t)kK1 * kB;         // [N*D]
  float* a2        = (float*)d_ws;                         // [2][B][D] fp32

  // a = x @ W1^T + b1 for both branches
  a2_kernel<<<kB, 256, 0, stream>>>(v1, v2, mt_w1, mt_b1, mts_w1, mts_b1, a2);

  // new_memory = memory (base image), then momentum-update the 64 rows in y
  int n4 = kN * kD / 4;
  copy_mem_kernel<<<(n4 + 255) / 256, 256, 0, stream>>>((const float4*)mem,
                                                        (float4*)out_mem, n4);
  update_mem_kernel<<<kB, 32, 0, stream>>>(mem, v2, y, out_mem);

  // Fused gather + WMMA chain + score
  size_t smem = (size_t)SMEM_USH * sizeof(unsigned short);
  (void)hipFuncSetAttribute((const void*)fused_kernel,
                            hipFuncAttributeMaxDynamicSharedMemorySize, (int)smem);
  fused_kernel<<<512, NW * 32, smem, stream>>>(mem, idx, a2,
      mt_w2, mt_b2, mt_wv, mt_bv, ht_w, ht_b,
      mts_w2, mts_b2, mts_wv, mts_bv, hts_w, hts_b, out_score);
}
